// VQAttention_77309411362
// MI455X (gfx1250) — matrix operand order repaired and checked by reading, not compiled, not run on hardware
//
#include <hip/hip_runtime.h>

typedef __attribute__((ext_vector_type(16))) __bf16 v16bf;
typedef __attribute__((ext_vector_type(8)))  __bf16 v8bf;
typedef __attribute__((ext_vector_type(8)))  float  v8f;

constexpr int Bc  = 8;
constexpr int Tc  = 512;
constexpr int Dc  = 1024;
constexpr int Hc  = 8;
constexpr int DKc = 128;
constexpr int DVc = 128;
constexpr int Cc  = 512;
constexpr int Mc  = 512;
constexpr int Sc  = Mc + Tc;     // 1024 recent positions
constexpr int BT  = Bc * Tc;     // 4096 rows
constexpr int HD  = Hc * DKc;    // 1024 (= H*DV as well)

// ---------------- bf16 helpers ----------------
__device__ __forceinline__ __bf16 f2bf(float f) {
  union { float f; unsigned u; } v; v.f = f;
  unsigned r = v.u + 0x7FFFu + ((v.u >> 16) & 1u);
  union { unsigned short s; __bf16 b; } o; o.s = (unsigned short)(r >> 16);
  return o.b;
}

__device__ __forceinline__ v8f wmma_bf(v16bf a, v16bf b, v8f c) {
  // D = A(16xK=32) * B(32x16) + C, f32 accumulate
  return __builtin_amdgcn_wmma_f32_16x16x32_bf16(false, a, false, b, (short)0, c,
                                                 false, false);
}

// A-fragment (16x32, row per lane%16): lanes<16 load K {0..15}, lanes>=16 {16..31} (contiguous order, same as B)
__device__ __forceinline__ v16bf load_afrag(const __bf16* __restrict__ row, int k0, int lg) {
  v8bf lo = *(const v8bf*)(row + k0 + lg * 16);
  v8bf hi = *(const v8bf*)(row + k0 + lg * 16 + 8);
  return __builtin_shufflevector(lo, hi, 0,1,2,3,4,5,6,7,8,9,10,11,12,13,14,15);
}

// B-fragment (32x16, col per lane%16, Bt row-major NxK): contiguous 16 elements
__device__ __forceinline__ v16bf load_bfrag(const __bf16* __restrict__ row, int k0, int lg) {
  return *(const v16bf*)(row + k0 + lg * 16);
}

// ---------------- small prep kernels ----------------
__global__ void k_init(float* msum, float* outl) {
  if (threadIdx.x == 0) { msum[0] = 0.f; outl[0] = 0.f; outl[1] = 0.f; }
}

__global__ void k_masksum(const float* __restrict__ lm, float* msum) {
  __shared__ float sh[256];
  int tid = threadIdx.x;
  int i = blockIdx.x * 256 + tid;
  sh[tid] = (i < BT) ? lm[i] : 0.f;
  __syncthreads();
  for (int o = 128; o; o >>= 1) { if (tid < o) sh[tid] += sh[tid + o]; __syncthreads(); }
  if (tid == 0) atomicAdd(msum, sh[0]);
}

// dst[n*Ksz + k] = src[k*Nsz + n]  (fp32 -> bf16 transposed)
__global__ void k_transpose_bf(const float* __restrict__ src, __bf16* __restrict__ dst,
                               int Ksz, int Nsz) {
  size_t tot = (size_t)Ksz * Nsz;
  for (size_t i = blockIdx.x * (size_t)blockDim.x + threadIdx.x; i < tot;
       i += (size_t)gridDim.x * blockDim.x) {
    size_t n = i / Ksz, k = i % Ksz;
    dst[i] = f2bf(src[k * (size_t)Nsz + n]);
  }
}

__global__ void k_tobf(const float* __restrict__ src, __bf16* __restrict__ dst, size_t n) {
  for (size_t i = blockIdx.x * (size_t)blockDim.x + threadIdx.x; i < n;
       i += (size_t)gridDim.x * blockDim.x)
    dst[i] = f2bf(src[i]);
}

__global__ void k_c2(const float* __restrict__ cb, float* __restrict__ c2) {
  int i = blockIdx.x * blockDim.x + threadIdx.x; // h*C + c
  if (i < Hc * Cc) {
    const float* r = cb + (size_t)i * DKc;
    float s = 0.f;
    for (int d = 0; d < DKc; ++d) s += r[d] * r[d];
    c2[i] = s;
  }
}

__global__ void k_kcat_head(const float* __restrict__ xlk, __bf16* __restrict__ kcat) {
  size_t tot = (size_t)Bc * Hc * Mc * DKc;
  for (size_t i = blockIdx.x * (size_t)blockDim.x + threadIdx.x; i < tot;
       i += (size_t)gridDim.x * blockDim.x) {
    int d = (int)(i % DKc);
    size_t rest = i / DKc;
    int s = (int)(rest % Mc);
    size_t bh = rest / Mc;
    kcat[(bh * Sc + s) * DKc + d] = f2bf(xlk[i]);
  }
}

// vT[bh][d][s] : s<M from xl_v[bh][s][d] else from y_v[(b*T + s-M)][h*DV + d]
__global__ void k_fill_vT(const float* __restrict__ xlv, const float* __restrict__ yv,
                          __bf16* __restrict__ vT) {
  size_t tot = (size_t)Bc * Hc * DVc * Sc;
  for (size_t i = blockIdx.x * (size_t)blockDim.x + threadIdx.x; i < tot;
       i += (size_t)gridDim.x * blockDim.x) {
    int s = (int)(i % Sc);
    size_t rest = i / Sc;
    int d = (int)(rest % DVc);
    size_t bh = rest / DVc;
    float val;
    if (s < Mc) val = xlv[(bh * Mc + s) * DVc + d];
    else {
      size_t b = bh / Hc, h = bh % Hc;
      val = yv[(b * Tc + (size_t)(s - Mc)) * HD + h * DVc + d];
    }
    vT[i] = f2bf(val);
  }
}

// uT[bh][d][c] from agg_upper[bh][c][d]
__global__ void k_fill_uT(const float* __restrict__ up, __bf16* __restrict__ uT) {
  size_t tot = (size_t)Bc * Hc * DVc * Cc;
  for (size_t i = blockIdx.x * (size_t)blockDim.x + threadIdx.x; i < tot;
       i += (size_t)gridDim.x * blockDim.x) {
    int c = (int)(i % Cc);
    size_t rest = i / Cc;
    int d = (int)(rest % DVc);
    size_t bh = rest / DVc;
    uT[i] = f2bf(up[(bh * Cc + c) * DVc + d]);
  }
}

// ---------------- LayerNorm ----------------
__global__ void k_ln(const float* __restrict__ x, const float* __restrict__ gam,
                     const float* __restrict__ bet, __bf16* __restrict__ xt) {
  __shared__ float sh[256];
  int row = blockIdx.x, tid = threadIdx.x;
  const float* xr = x + (size_t)row * Dc;
  float s = 0.f, s2 = 0.f;
  for (int i = tid; i < Dc; i += 256) { float v = xr[i]; s += v; s2 += v * v; }
  sh[tid] = s; __syncthreads();
  for (int o = 128; o; o >>= 1) { if (tid < o) sh[tid] += sh[tid + o]; __syncthreads(); }
  float mu = sh[0] / Dc; __syncthreads();
  sh[tid] = s2; __syncthreads();
  for (int o = 128; o; o >>= 1) { if (tid < o) sh[tid] += sh[tid + o]; __syncthreads(); }
  float var = sh[0] / Dc - mu * mu;
  float rs = rsqrtf(var + 1e-6f);
  for (int i = tid; i < Dc; i += 256)
    xt[(size_t)row * Dc + i] = f2bf((xr[i] - mu) * rs * gam[i] + bet[i]);
}

// ---------------- generic NT GEMM: C = A(MrxK) * Bt(NxK)^T ----------------
__global__ void __launch_bounds__(256)
k_gemm_nt(const __bf16* __restrict__ A, const __bf16* __restrict__ Bt,
          float* __restrict__ Cm, int Mr, int N, int K,
          long sA, long sB, int bModB, long sC) {
  int lane = threadIdx.x & 31, wave = threadIdx.x >> 5;
  int lg = lane >> 4, ln16 = lane & 15;
  int m0 = (blockIdx.y * 8 + wave) * 16;
  int n0 = blockIdx.x * 64;
  int z = blockIdx.z;
  const __bf16* Ab = A + (size_t)z * sA;
  const __bf16* Bb = Bt + (size_t)(z % bModB) * sB;
  float* Cb = Cm + (size_t)z * sC;
  if (m0 >= Mr) return;
  v8f zero = {0.f,0.f,0.f,0.f,0.f,0.f,0.f,0.f};
  v8f acc[4] = {zero, zero, zero, zero};
  const __bf16* arow = Ab + (size_t)(m0 + ln16) * K;
  for (int k0 = 0; k0 < K; k0 += 32) {
    v16bf af = load_afrag(arow, k0, lg);
#pragma unroll
    for (int j = 0; j < 4; ++j) {
      const __bf16* brow = Bb + (size_t)(n0 + j * 16 + ln16) * K;
      v16bf bfv = load_bfrag(brow, k0, lg);
      acc[j] = wmma_bf(af, bfv, acc[j]);
    }
  }
#pragma unroll
  for (int j = 0; j < 4; ++j)
#pragma unroll
    for (int r = 0; r < 8; ++r)
      Cb[(size_t)(m0 + 8 * lg + r) * N + n0 + j * 16 + ln16] = acc[j][r];
}

// ---------------- head-LN epilogue for q/k ----------------
__device__ __forceinline__ float blockReduceSum128(float v, float* sh) {
  int tid = threadIdx.x;
  sh[tid] = v; __syncthreads();
  for (int o = 64; o; o >>= 1) { if (tid < o) sh[tid] += sh[tid + o]; __syncthreads(); }
  float r = sh[0]; __syncthreads();
  return r;
}

__global__ void k_hln(const float* __restrict__ yq, const float* __restrict__ yk,
                      const float* __restrict__ xu,
                      __bf16* __restrict__ qbb, __bf16* __restrict__ kbf,
                      float* __restrict__ kf) {
  __shared__ float sh[128];
  int tid = threadIdx.x;
  int h = blockIdx.x % Hc;
  int bt = blockIdx.x / Hc;   // b*T + t
  size_t src = (size_t)bt * HD + h * DKc + tid;
  float qv = yq[src], kv = yk[src];
  float qs  = blockReduceSum128(qv, sh);
  float qs2 = blockReduceSum128(qv * qv, sh);
  float muq = qs / DKc, vq = qs2 / DKc - muq * muq, rq = rsqrtf(vq + 1e-6f);
  float ks  = blockReduceSum128(kv, sh);
  float ks2 = blockReduceSum128(kv * kv, sh);
  float muk = ks / DKc, vk = ks2 / DKc - muk * muk, rk = rsqrtf(vk + 1e-6f);
  int b = bt / Tc, t = bt % Tc;
  size_t dsto = (((size_t)b * Hc + h) * Tc + t) * DKc + tid;
  qbb[dsto] = f2bf((qv - muq) * rq + xu[h * DKc + tid]);
  float kn = (kv - muk) * rk;
  kbf[dsto] = f2bf(kn);
  kf[dsto] = kn;
}

// ---------------- VQ argmin + losses + quantized key write ----------------
__global__ void k_vq(const float* __restrict__ kc, const float* __restrict__ c2,
                     const float* __restrict__ cb, const float* __restrict__ kf,
                     const float* __restrict__ lm,
                     __bf16* __restrict__ kcat, float* __restrict__ outl) {
  __shared__ float bv[128];
  __shared__ int bix[128];
  __shared__ int zsh;
  int tid = threadIdx.x;
  int bh = blockIdx.x / Tc, t = blockIdx.x % Tc;
  int b = bh / Hc, h = bh % Hc;
  const float* kcr = kc + ((size_t)bh * Tc + t) * Cc;
  float best = 3.0e38f; int bi = 0;
  for (int c = tid; c < Cc; c += 128) {
    float dv = c2[h * Cc + c] - 2.f * kcr[c]; // k^2 term constant per row
    if (dv < best) { best = dv; bi = c; }
  }
  bv[tid] = best; bix[tid] = bi; __syncthreads();
  for (int o = 64; o; o >>= 1) {
    if (tid < o) {
      if (bv[tid + o] < bv[tid] ||
          (bv[tid + o] == bv[tid] && bix[tid + o] < bix[tid])) {
        bv[tid] = bv[tid + o]; bix[tid] = bix[tid + o];
      }
    }
    __syncthreads();
  }
  if (tid == 0) zsh = bix[0];
  __syncthreads();
  int z = zsh;
  float cv = cb[((size_t)h * Cc + z) * DKc + tid];
  kcat[((size_t)bh * Sc + Mc + t) * DKc + tid] = f2bf(cv);
  float kvv = kf[((size_t)bh * Tc + t) * DKc + tid];
  float df = kvv - cv; df *= df;
  __syncthreads();
  bv[tid] = df; __syncthreads();
  for (int o = 64; o; o >>= 1) { if (tid < o) bv[tid] += bv[tid + o]; __syncthreads(); }
  if (tid == 0) {
    float wgt = lm[b * Tc + t];
    atomicAdd(&outl[0], bv[0] * wgt);
    atomicAdd(&outl[1], bv[0] * wgt);
  }
}

// ---------------- flash attention block step ----------------
__device__ __forceinline__ void attn_block(
    const __bf16* __restrict__ Abase, const v16bf* qf,
    int sBase, int tglob, int maskM,
    const float* __restrict__ lower,
    const __bf16* __restrict__ Vbase, int ldV,
    int lg, int ln16, float& mrun, float& den, v8f* num) {
  constexpr float inv_tau = 0.088388347648318447f; // 1/sqrt(DK)
  v8f st[2];
#pragma unroll
  for (int ti = 0; ti < 2; ++ti) {
    v8f acc = {0.f,0.f,0.f,0.f,0.f,0.f,0.f,0.f};
    const __bf16* arow = Abase + (size_t)(sBase + ti * 16 + ln16) * DKc;
#pragma unroll
    for (int c = 0; c < 4; ++c) {
      v16bf af = load_afrag(arow, 32 * c, lg);
      acc = wmma_bf(af, qf[c], acc);   // sT tile: lanes=t, regs=s
    }
    st[ti] = acc;
  }
  float pv[16];
  float bm = -3.0e38f;
#pragma unroll
  for (int ti = 0; ti < 2; ++ti)
#pragma unroll
    for (int r = 0; r < 8; ++r) {
      float v = st[ti][r] * inv_tau;
      int s = sBase + ti * 16 + 8 * lg + r;
      if (maskM >= 0 && (s - maskM) > tglob) v = -1.0e30f;
      pv[ti * 8 + r] = v;
      bm = fmaxf(bm, v);
    }
  bm = fmaxf(bm, __shfl_xor(bm, 16, 32));
  float mnew = fmaxf(mrun, bm);
  float scale = __expf(mrun - mnew);
  v16bf pf;
  float ls = 0.f;
#pragma unroll
  for (int ti = 0; ti < 2; ++ti)
#pragma unroll
    for (int r = 0; r < 8; ++r) {
      int s = sBase + ti * 16 + 8 * lg + r;
      float p = __expf(pv[ti * 8 + r] - mnew);
      float wgt = lower ? lower[s] : 1.0f;
      ls += p * wgt;
      pf[ti * 8 + r] = f2bf(p);  // exactly the A-frag layout for P (16t x 32s)
    }
  ls += __shfl_xor(ls, 16, 32);
  den = den * scale + ls;
  float s8[8];
#pragma unroll
  for (int r = 0; r < 8; ++r) s8[r] = __shfl(scale, 8 * lg + r, 32);
#pragma unroll
  for (int dt = 0; dt < 8; ++dt) {
#pragma unroll
    for (int r = 0; r < 8; ++r) num[dt][r] *= s8[r];
    const __bf16* vrow = Vbase + (size_t)(dt * 16 + ln16) * ldV + sBase;
    v8bf vlo = *(const v8bf*)(vrow + lg * 8);
    v8bf vhi = *(const v8bf*)(vrow + 16 + lg * 8);
    v16bf bfv = __builtin_shufflevector(vlo, vhi, 0,1,2,3,4,5,6,7,8,9,10,11,12,13,14,15);
    num[dt] = wmma_bf(pf, bfv, num[dt]);
  }
  mrun = mnew;
}

__global__ void __launch_bounds__(256)
k_attn(const __bf16* __restrict__ qbb, const __bf16* __restrict__ kcat,
       const __bf16* __restrict__ vT, const __bf16* __restrict__ cbbf,
       const __bf16* __restrict__ uT, const float* __restrict__ lower,
       const float* __restrict__ yg, __bf16* __restrict__ obf) {
  int bh = blockIdx.x;
  int b = bh / Hc, h = bh % Hc;
  int lane = threadIdx.x & 31, wave = threadIdx.x >> 5;
  int lg = lane >> 4, ln16 = lane & 15;
  int t0 = blockIdx.y * 8 + wave;      // 16-row tile index, 0..31
  int tglob = t0 * 16 + ln16;

  const __bf16* qrow = qbb + ((size_t)bh * Tc + tglob) * DKc;
  v16bf qf[4];
#pragma unroll
  for (int c = 0; c < 4; ++c) qf[c] = *(const v16bf*)(qrow + 32 * c + lg * 16);

  v8f zero = {0.f,0.f,0.f,0.f,0.f,0.f,0.f,0.f};
  v8f num[8];
#pragma unroll
  for (int i = 0; i < 8; ++i) num[i] = zero;
  float mrun = -3.0e38f, den = 0.f;

  // recent (causal-masked) part
  const __bf16* Krec = kcat + (size_t)bh * Sc * DKc;
  const __bf16* Vrec = vT + (size_t)bh * DVc * Sc;
  int nrec = (Mc + 16 * (t0 + 1) + 31) / 32;
  for (int sb = 0; sb < nrec; ++sb)
    attn_block(Krec, qf, sb * 32, tglob, Mc, nullptr, Vrec, Sc, lg, ln16, mrun, den, num);

  // aggregate (codebook) part, lower-weighted denominator
  const __bf16* Kagg = cbbf + (size_t)h * Cc * DKc;
  const __bf16* Vagg = uT + (size_t)bh * DVc * Cc;
  const float* lw = lower + (size_t)bh * Cc;
  for (int sb = 0; sb < Cc / 32; ++sb)
    attn_block(Kagg, qf, sb * 32, tglob, -1, lw, Vagg, Cc, lg, ln16, mrun, den, num);

  float d8[8];
#pragma unroll
  for (int r = 0; r < 8; ++r) d8[r] = __shfl(den, 8 * lg + r, 32);
#pragma unroll
  for (int dt = 0; dt < 8; ++dt) {
    int dcol = dt * 16 + ln16;
#pragma unroll
    for (int r = 0; r < 8; ++r) {
      int tg = t0 * 16 + 8 * lg + r;
      float wv = num[dt][r] / d8[r];
      size_t gi = ((size_t)(b * Tc + tg)) * HD + h * DVc + dcol;
      float gv = yg[gi];
      float sil = gv / (1.f + __expf(-gv));
      obf[gi] = f2bf(wv * sil);
    }
  }
}

__global__ void k_final(const float* __restrict__ msum, float* __restrict__ outl) {
  if (threadIdx.x == 0) {
    float denom = msum[0] * (float)Hc + 1e-6f;
    outl[0] /= denom;
    outl[1] /= denom;
  }
}

// ---------------- host ----------------
extern "C" void kernel_launch(void* const* d_in, const int* in_sizes, int n_in,
                              void* d_out, int out_size, void* d_ws, size_t ws_size,
                              hipStream_t stream) {
  (void)in_sizes; (void)n_in; (void)out_size; (void)ws_size;
  const float* x    = (const float*)d_in[0];
  const float* lmask= (const float*)d_in[2];
  const float* xlk  = (const float*)d_in[3];
  const float* xlv  = (const float*)d_in[4];
  const float* aggU = (const float*)d_in[5];
  const float* aggL = (const float*)d_in[6];
  const float* lng  = (const float*)d_in[7];
  const float* lnb  = (const float*)d_in[8];
  const float* wq   = (const float*)d_in[9];
  const float* wk   = (const float*)d_in[10];
  const float* wv   = (const float*)d_in[11];
  const float* wg   = (const float*)d_in[12];
  const float* wres = (const float*)d_in[13];
  const float* xu   = (const float*)d_in[14];
  const float* cb   = (const float*)d_in[15];
  float* out = (float*)d_out;

  char* wp = (char*)d_ws;
  auto carve = [&](size_t bytes) -> char* {
    char* p = wp; wp += (bytes + 255) & ~(size_t)255; return p;
  };
  __bf16* xtbf = (__bf16*)carve((size_t)BT * Dc * 2);
  __bf16* wqT  = (__bf16*)carve((size_t)HD * Dc * 2);
  __bf16* wkT  = (__bf16*)carve((size_t)HD * Dc * 2);
  __bf16* wvT  = (__bf16*)carve((size_t)HD * Dc * 2);
  __bf16* wgT  = (__bf16*)carve((size_t)HD * Dc * 2);
  __bf16* wrT  = (__bf16*)carve((size_t)Dc * HD * 2);
  __bf16* cbbf = (__bf16*)carve((size_t)Hc * Cc * DKc * 2);
  float*  c2   = (float*) carve((size_t)Hc * Cc * 4);
  float*  yq   = (float*) carve((size_t)BT * HD * 4);
  float*  yk   = (float*) carve((size_t)BT * HD * 4);
  float*  yv   = (float*) carve((size_t)BT * HD * 4);
  float*  yg   = (float*) carve((size_t)BT * HD * 4);
  __bf16* qbb  = (__bf16*)carve((size_t)Bc * Hc * Tc * DKc * 2);
  __bf16* kbf  = (__bf16*)carve((size_t)Bc * Hc * Tc * DKc * 2);
  float*  kf   = (float*) carve((size_t)Bc * Hc * Tc * DKc * 4);
  __bf16* kcat = (__bf16*)carve((size_t)Bc * Hc * Sc * DKc * 2);
  __bf16* vT   = (__bf16*)carve((size_t)Bc * Hc * DVc * Sc * 2);
  __bf16* uT   = (__bf16*)carve((size_t)Bc * Hc * DVc * Cc * 2);
  float*  kc   = (float*) carve((size_t)Bc * Hc * Tc * Cc * 4);
  __bf16* obf  = (__bf16*)carve((size_t)BT * HD * 2);
  float*  msum = (float*) carve(256);

  float* outl = out + (size_t)BT * Dc;

  k_init<<<1, 32, 0, stream>>>(msum, outl);
  k_masksum<<<BT / 256, 256, 0, stream>>>(lmask, msum);

  k_transpose_bf<<<1024, 256, 0, stream>>>(wq, wqT, Dc, HD);
  k_transpose_bf<<<1024, 256, 0, stream>>>(wk, wkT, Dc, HD);
  k_transpose_bf<<<1024, 256, 0, stream>>>(wv, wvT, Dc, HD);
  k_transpose_bf<<<1024, 256, 0, stream>>>(wg, wgT, Dc, HD);
  k_transpose_bf<<<1024, 256, 0, stream>>>(wres, wrT, HD, Dc);
  k_tobf<<<1024, 256, 0, stream>>>(cb, cbbf, (size_t)Hc * Cc * DKc);
  k_c2<<<(Hc * Cc + 255) / 256, 256, 0, stream>>>(cb, c2);
  k_kcat_head<<<2048, 256, 0, stream>>>(xlk, kcat);

  k_ln<<<BT, 256, 0, stream>>>(x, lng, lnb, xtbf);

  dim3 gp(HD / 64, BT / 128, 1);
  k_gemm_nt<<<gp, 256, 0, stream>>>(xtbf, wqT, yq, BT, HD, Dc, 0, 0, 1, 0);
  k_gemm_nt<<<gp, 256, 0, stream>>>(xtbf, wkT, yk, BT, HD, Dc, 0, 0, 1, 0);
  k_gemm_nt<<<gp, 256, 0, stream>>>(xtbf, wvT, yv, BT, HD, Dc, 0, 0, 1, 0);
  k_gemm_nt<<<gp, 256, 0, stream>>>(xtbf, wgT, yg, BT, HD, Dc, 0, 0, 1, 0);

  k_hln<<<BT * Hc, 128, 0, stream>>>(yq, yk, xu, qbb, kbf, kf);

  dim3 gkc(Cc / 64, Tc / 128, Bc * Hc);
  k_gemm_nt<<<gkc, 256, 0, stream>>>(kbf, cbbf, kc, Tc, Cc, DKc,
                                     (long)Tc * DKc, (long)Cc * DKc, Hc,
                                     (long)Tc * Cc);

  k_vq<<<Bc * Hc * Tc, 128, 0, stream>>>(kc, c2, cb, kf, lmask, kcat, outl);

  k_fill_vT<<<4096, 256, 0, stream>>>(xlv, yv, vT);
  k_fill_uT<<<4096, 256, 0, stream>>>(aggU, uT);

  dim3 ga(Bc * Hc, Tc / 128, 1);
  k_attn<<<ga, 256, 0, stream>>>(qbb, kcat, vT, cbbf, uT, aggL, yg, obf);

  dim3 gr(Dc / 64, BT / 128, 1);
  k_gemm_nt<<<gr, 256, 0, stream>>>(obf, wrT, out, BT, Dc, HD, 0, 0, 1, 0);

  k_final<<<1, 32, 0, stream>>>(msum, outl);
}
